// HVQ_TR_switch_OT_15745350108066
// MI455X (gfx1250) — matrix operand order repaired and checked by reading, not compiled, not run on hardware
//
#include <hip/hip_runtime.h>

typedef __attribute__((ext_vector_type(16))) __bf16 v16bf;
typedef __attribute__((ext_vector_type(8)))  float  v8f;
typedef __attribute__((ext_vector_type(4)))  float  fvec4;

#define NB 128      // batch
#define LL 196      // tokens per batch
#define NE 512      // codebook size
#define EPSF 1e-8f

// flat output layout (return order: quant_list, diff, id_list, ot_score)
constexpr size_t QUANT_SZ = 4ull * NB * LL * 128;  // 12,845,056
constexpr size_t DIFF_OFF = QUANT_SZ;
constexpr size_t ID_OFF   = DIFF_OFF + 1;
constexpr size_t ID_SZ    = 4ull * NB * LL;        // 100,352
constexpr size_t OT_OFF   = ID_OFF + ID_SZ;

__device__ __forceinline__ unsigned f2bfbits(float f) {
  union { float f; unsigned u; } a; a.f = f;
  return (a.u + 0x7FFFu + ((a.u >> 16) & 1u)) >> 16;   // round-to-nearest-even
}
__device__ __forceinline__ __bf16 f2bf(float f) {
  unsigned short h = (unsigned short)f2bfbits(f);
  return __builtin_bit_cast(__bf16, h);
}

// fragment-order index for element (kk in 0..31, n in 0..511) of a K-slab:
// lane = ((kk/8)%2)*16 + n%16, tile = n/16, half = (kk/16)*8 + kk%8
__device__ __forceinline__ int frag_idx(int kk, int n) {
  return (((n >> 4) * 32 + ((kk >> 3) & 1) * 16 + (n & 15)) << 4) + (((kk >> 4) << 3) | (kk & 7));
}

__global__ void init_diff(float* out) { out[DIFF_OFF] = 0.0f; }

// Precompute ||e_k||^2 for every (codebook, class, code) column.
__global__ void norms_kernel(const float* __restrict__ cb_top,
                             const float* __restrict__ cb_lvl,
                             float* __restrict__ en_top,
                             float* __restrict__ en_lvl) {
  int t = blockIdx.x * blockDim.x + threadIdx.x;
  if (t < 15 * NE) {
    int n = t & (NE - 1);
    const float* e = cb_top + (size_t)(t >> 9) * 64 * NE;
    float s = 0.f;
    for (int k = 0; k < 64; ++k) { float v = e[(size_t)k * NE + n]; s += v * v; }
    en_top[t] = s;
  } else if (t < 15 * NE + 60 * NE) {
    int t2 = t - 15 * NE;
    int n = t2 & (NE - 1);
    const float* e = cb_lvl + (size_t)(t2 >> 9) * 128 * NE;
    float s = 0.f;
    for (int k = 0; k < 128; ++k) { float v = e[(size_t)k * NE + n]; s += v * v; }
    en_lvl[t2] = s;
  }
}

// One-time pre-pass: convert codebooks to bf16 in WMMA B-fragment order.
// Layout per class: [kstep][(ntile*32 + lane)*16 + half]  (32KB per kstep).
__global__ void pack_cb(const float* __restrict__ cb_top,
                        const float* __restrict__ cb_lvl,
                        __bf16* __restrict__ pk_top,
                        __bf16* __restrict__ pk_lvl) {
  size_t t = (size_t)blockIdx.x * blockDim.x + threadIdx.x;
  const size_t TOPN = 15ull * 64 * NE;
  if (t < TOPN) {
    size_t cls = t / (64 * NE);
    int rem = (int)(t - cls * (64 * NE));
    int k = rem >> 9, n = rem & (NE - 1);
    pk_top[cls * (64 * NE) + (size_t)(k >> 5) * (32 * NE) + frag_idx(k & 31, n)]
        = f2bf(cb_top[t]);
  } else {
    size_t t2 = t - TOPN;
    if (t2 < 60ull * 128 * NE) {
      size_t cls = t2 / (128 * NE);
      int rem = (int)(t2 - cls * (128 * NE));
      int k = rem >> 9, n = rem & (NE - 1);
      pk_lvl[cls * (128 * NE) + (size_t)(k >> 5) * (32 * NE) + frag_idx(k & 31, n)]
          = f2bf(cb_lvl[t2]);
    }
  }
}

// One block = one (m-tile, batch). 8 waves; each wave owns 4 N-tiles of 16.
// E-slab staging is a pure 32KB contiguous bf16 copy (pre-packed in ws);
// each A/B WMMA operand is one 32B LDS read.
template<int KDIM, bool IS_TOP, bool IS_OT>
__global__ __launch_bounds__(256) void vq_kernel(
    const float* __restrict__ input, const float* __restrict__ cb_top,
    const float* __restrict__ cb_lvl, const int* __restrict__ label,
    float* __restrict__ out, float* __restrict__ q4ws,
    const float* __restrict__ en_top, const float* __restrict__ en_lvl,
    const __bf16* __restrict__ pk_top, const __bf16* __restrict__ pk_lvl,
    int j)
{
  // E fragments (32 n-tiles x 32 lanes x 16 bf16 = 32KB) aliased with the
  // f32 dot buffer (16 x 512 = 32KB): slab is dead once accumulators spill.
  __shared__ __align__(32) char slabmem[32 * NE * 2];
  __shared__ __align__(32) __bf16 xfrag[KDIM * 16];   // (K/32) ksteps x 32 lanes x 16 bf16
  __shared__ __align__(16) float  enr[NE];
  __shared__ float  xn2s[16];
  __shared__ int    indb[16];
  __shared__ float  red[256];

  __bf16* eslab  = (__bf16*)slabmem;
  float*  dotbuf = (float*)slabmem;

  const int mtile = blockIdx.x;
  const int b     = blockIdx.y;
  const int lbl   = label[b];
  const int tid   = threadIdx.x;
  const int wave  = tid >> 5, lane = tid & 31;
  const int lmod  = lane & 15, lgrp = lane >> 4;
  const int row   = tid >> 4, sub  = tid & 15;   // 16 threads per row

  const float* emb = IS_TOP ? cb_top + (size_t)lbl * 64 * NE
                            : cb_lvl + ((size_t)j * 15 + lbl) * 128 * NE;
  const float* enp = IS_TOP ? en_top + (size_t)lbl * NE
                            : en_lvl + ((size_t)j * 15 + lbl) * NE;
  const __bf16* pke = IS_TOP ? pk_top + (size_t)lbl * 64 * NE
                             : pk_lvl + ((size_t)j * 15 + lbl) * 128 * NE;

  // ---- stage x tile into A-fragment order (bf16) + f32 row norms ----
  {
    int m  = mtile * 16 + row;
    int mc = m < LL ? m : LL - 1;               // clamp padded rows
    float s2 = 0.f;
    for (int k = sub; k < KDIM; k += 16) {
      float xv;
      if (IS_TOP) {
        xv = input[(((size_t)4 * NB + b) * LL + mc) * 64 + k];
      } else {
        xv = (k < 64) ? input[(((size_t)j * NB + b) * LL + mc) * 64 + k]
                      : q4ws[((size_t)b * LL + mc) * 64 + (k - 64)];
      }
      s2 += xv * xv;
      // inverse of ISA 7.12.2 16-bit A-fragment K mapping
      int kb  = k >> 5, kk = k & 31;
      int hi4 = (kk >> 4) << 2;
      int rem = kk & 15, lg = rem >> 3, r8 = rem & 7;
      int h   = (((hi4 + (r8 >> 1)) << 1) | (r8 & 1));
      xfrag[((kb * 32 + lg * 16 + row) << 4) + h] = f2bf(xv);
    }
    for (int mk = 8; mk; mk >>= 1) s2 += __shfl_xor(s2, mk, 32);
    if (sub == 0) xn2s[row] = s2;
  }
  for (int n = tid; n < NE; n += 256) enr[n] = enp[n];
  __syncthreads();

  // ---- GEMM via WMMA: dot[16][512] ----
  v8f acc[4] = {};
  const v16bf* xf = (const v16bf*)xfrag;
  const v16bf* ef = (const v16bf*)eslab;
  v16bf* edst = (v16bf*)eslab;

  for (int kb = 0; kb < KDIM; kb += 32) {
    // stage E-slab: pure contiguous copy of pre-packed fragments (32KB)
    const v16bf* src = (const v16bf*)(pke + (size_t)(kb >> 5) * (32 * NE));
#pragma unroll
    for (int i = 0; i < 4; ++i)
      edst[tid + i * 256] = src[tid + i * 256];
    if (kb + 32 < KDIM)
      __builtin_prefetch((const void*)(src + 1024 + tid * 4), 0, 0);
    __syncthreads();

    const int ks = kb >> 5;
    v16bf afrag = xf[ks * 32 + lane];                 // two ds_load_b128
#pragma unroll
    for (int t = 0; t < 4; ++t) {
      v16bf bfrag = ef[(wave * 4 + t) * 32 + lane];   // two ds_load_b128
      acc[t] = __builtin_amdgcn_wmma_f32_16x16x32_bf16(
          false, afrag, false, bfrag, (short)0, acc[t], false, false);
    }
    __syncthreads();
  }

  // ---- spill accumulators to LDS dot buffer (aliases E-slab) ----
#pragma unroll
  for (int t = 0; t < 4; ++t) {
    int ncol = (wave * 4 + t) * 16 + lmod;
#pragma unroll
    for (int r = 0; r < 8; ++r)
      dotbuf[(r + lgrp * 8) * NE + ncol] = acc[t][r];   // C layout: VGPR r -> M=r+8*grp
  }
  __syncthreads();

  // ---- argmin over 512 codes (first-min tie-break, like jnp.argmin) ----
  const int m = mtile * 16 + row;
  const float xn2 = xn2s[row];
  const fvec4* dp = (const fvec4*)(dotbuf + row * NE + sub * 32);
  const fvec4* ep = (const fvec4*)(enr + sub * 32);
  float bestv = 3.4e38f; int besti = 0;
#pragma unroll
  for (int i = 0; i < 8; ++i) {
    fvec4 d4 = dp[i], e4 = ep[i];
#pragma unroll
    for (int c = 0; c < 4; ++c) {
      float d = xn2 - 2.0f * d4[c] + e4[c];
      int n = sub * 32 + i * 4 + c;
      if (d < bestv) { bestv = d; besti = n; }
    }
  }
  for (int mk = 8; mk; mk >>= 1) {
    float ov = __shfl_xor(bestv, mk, 32);
    int   oi = __shfl_xor(besti, mk, 32);
    if (ov < bestv || (ov == bestv && oi < besti)) { bestv = ov; besti = oi; }
  }
  if (sub == 0) {
    indb[row] = besti;
    if (!IS_TOP && m < LL)
      out[ID_OFF + ((size_t)(3 - j) * NB + b) * LL + m] = (float)besti;
  }

  // ---- OT score (level 0): reuse dot for cos-sim and softmax(dot/512) ----
  if (IS_OT) {
    float mx = -3.4e38f;
#pragma unroll
    for (int i = 0; i < 8; ++i) {
      fvec4 d4 = dp[i];
#pragma unroll
      for (int c = 0; c < 4; ++c) mx = fmaxf(mx, d4[c]);
    }
    for (int mk = 8; mk; mk >>= 1) mx = fmaxf(mx, __shfl_xor(mx, mk, 32));
    mx *= (1.0f / NE);
    const float nx = sqrtf(xn2) + EPSF;
    float sw = 0.f, swc = 0.f;
#pragma unroll
    for (int i = 0; i < 8; ++i) {
      fvec4 d4 = dp[i], e4 = ep[i];
#pragma unroll
      for (int c = 0; c < 4; ++c) {
        float dt = d4[c];
        float w  = __expf(dt * (1.0f / NE) - mx);
        float cd = 1.0f - dt / (nx * (sqrtf(e4[c]) + EPSF));
        sw += w; swc += w * cd;
      }
    }
    for (int mk = 8; mk; mk >>= 1) {
      sw  += __shfl_xor(sw,  mk, 32);
      swc += __shfl_xor(swc, mk, 32);
    }
    if (sub == 0 && m < LL)
      out[OT_OFF + (size_t)b * LL + m] = swc / sw;
  }
  __syncthreads();

  // ---- gather q = E[:, ind], write quant (or q4 to ws), accumulate diff ----
  float dsum = 0.f;
  if (m < LL) {
    int ind = indb[row];
    for (int k = sub; k < KDIM; k += 16) {
      float q = emb[(size_t)k * NE + ind];
      float xo;
      if (IS_TOP) xo = input[(((size_t)4 * NB + b) * LL + m) * 64 + k];
      else        xo = (k < 64) ? input[(((size_t)j * NB + b) * LL + m) * 64 + k]
                                : q4ws[((size_t)b * LL + m) * 64 + (k - 64)];
      float df = q - xo; dsum += df * df;
      if (IS_TOP) q4ws[((size_t)b * LL + m) * 64 + k] = q;
      else        out[(((size_t)(3 - j) * NB + b) * LL + m) * 128 + k] = q;
    }
  }
  red[tid] = dsum;
  __syncthreads();
  for (int s2 = 128; s2; s2 >>= 1) {
    if (tid < s2) red[tid] += red[tid + s2];
    __syncthreads();
  }
  if (tid == 0) atomicAdd(out + DIFF_OFF, red[0] * (1.0f / (LL * KDIM)));
}

extern "C" void kernel_launch(void* const* d_in, const int* in_sizes, int n_in,
                              void* d_out, int out_size, void* d_ws, size_t ws_size,
                              hipStream_t stream) {
  (void)in_sizes; (void)n_in; (void)out_size; (void)ws_size;
  const float* input = (const float*)d_in[0];   // (5,128,196,64)
  const float* cbtop = (const float*)d_in[1];   // (15,64,512)
  const float* cblvl = (const float*)d_in[2];   // (4,15,128,512)
  const int*   label = (const int*)d_in[3];     // (128,)
  float* out = (float*)d_out;

  float* q4ws   = (float*)d_ws;                       // 128*196*64 f32
  float* en_top = q4ws + (size_t)NB * LL * 64;        // 15*512 f32
  float* en_lvl = en_top + (size_t)15 * NE;           // 60*512 f32
  __bf16* pk_top = (__bf16*)(en_lvl + (size_t)60 * NE); // 15*64*512 bf16
  __bf16* pk_lvl = pk_top + 15ull * 64 * NE;            // 60*128*512 bf16

  init_diff<<<dim3(1), dim3(1), 0, stream>>>(out);
  norms_kernel<<<dim3(150), dim3(256), 0, stream>>>(cbtop, cblvl, en_top, en_lvl);
  // 15*64*512 + 60*128*512 = 4,423,680 elements
  pack_cb<<<dim3(17280), dim3(256), 0, stream>>>(cbtop, cblvl, pk_top, pk_lvl);

  dim3 grid(13, 128);  // 13 M-tiles of 16 rows x 128 batches
  vq_kernel<64,  true,  false><<<grid, 256, 0, stream>>>(input, cbtop, cblvl, label, out, q4ws, en_top, en_lvl, pk_top, pk_lvl, 4);
  vq_kernel<128, false, false><<<grid, 256, 0, stream>>>(input, cbtop, cblvl, label, out, q4ws, en_top, en_lvl, pk_top, pk_lvl, 3);
  vq_kernel<128, false, false><<<grid, 256, 0, stream>>>(input, cbtop, cblvl, label, out, q4ws, en_top, en_lvl, pk_top, pk_lvl, 2);
  vq_kernel<128, false, false><<<grid, 256, 0, stream>>>(input, cbtop, cblvl, label, out, q4ws, en_top, en_lvl, pk_top, pk_lvl, 1);
  vq_kernel<128, false, true ><<<grid, 256, 0, stream>>>(input, cbtop, cblvl, label, out, q4ws, en_top, en_lvl, pk_top, pk_lvl, 0);
}
